// ResidualVectorQuantization_46007689674756
// MI455X (gfx1250) — matrix operand order repaired and checked by reading, not compiled, not run on hardware
//
#include <hip/hip_runtime.h>
#include <hip/hip_bf16.h>

typedef __attribute__((ext_vector_type(16))) __bf16 v16bf;
typedef __attribute__((ext_vector_type(8)))  float  v8f;

#define RVQ_B 65536
#define RVQ_D 256
#define RVQ_Q 4
#define RVQ_K 1024
#define ROWS  32                  // rows per block (2 WMMA A-tiles)
#define NBLK  (RVQ_B / ROWS)      // 2048 M-blocks

// ---------------------------------------------------------------------------
// codebook fp32 -> bf16 hi/lo split
// ---------------------------------------------------------------------------
__global__ void rvq_convert_kernel(const float* __restrict__ cb,
                                   __bf16* __restrict__ whi,
                                   __bf16* __restrict__ wlo) {
    size_t i = (size_t)blockIdx.x * blockDim.x + threadIdx.x;  // Q*K*D elements
    float w = cb[i];
    __bf16 h = (__bf16)w;
    whi[i] = h;
    wlo[i] = (__bf16)(w - (float)h);
}

// ---------------------------------------------------------------------------
// c2[code] = ||W_code||^2  over all Q*K codes (one wave per code, coalesced)
// ---------------------------------------------------------------------------
__global__ void rvq_c2_kernel(const float* __restrict__ cb, float* __restrict__ c2) {
    int wv = threadIdx.x >> 5, lane = threadIdx.x & 31;
    int code = blockIdx.x * 4 + wv;          // grid = (Q*K)/4 blocks of 128
    const float* p = cb + (size_t)code * RVQ_D;
    float s = 0.f;
    #pragma unroll
    for (int d = 0; d < RVQ_D / 32; ++d) {
        float v = p[lane + d * 32];
        s += v * v;
    }
    #pragma unroll
    for (int off = 16; off >= 1; off >>= 1) s += __shfl_xor(s, off, 32);
    if (lane == 0) c2[code] = s;
}

// ---------------------------------------------------------------------------
// build a 16x32 bf16 A fragment (lane half h holds contiguous k = 16h..16h+15) out of LDS
// ---------------------------------------------------------------------------
__device__ __forceinline__ v16bf load_afrag(const __bf16* __restrict__ p,
                                            int o1, int o2) {
    v16bf a;
    #pragma unroll
    for (int i = 0; i < 8; ++i) {
        a[i]     = p[o1 + i];
        a[8 + i] = p[o2 + i];
    }
    return a;
}

// ---------------------------------------------------------------------------
// One RVQ stage: distances via bf16-split WMMA, argmin, gather, residual update
// grid = 2048 blocks (32 rows each), block = 128 threads (4 waves)
// Each wave sweeps 16 N-tiles; each loaded B fragment feeds 2 A-tiles (6 wmma)
// ---------------------------------------------------------------------------
__global__ void __launch_bounds__(128)
rvq_stage_kernel(const float* __restrict__ resin, float* __restrict__ resout,
                 const float* __restrict__ cb,
                 const __bf16* __restrict__ whi, const __bf16* __restrict__ wlo,
                 const float* __restrict__ c2, float* __restrict__ out,
                 float* __restrict__ partials, int stage) {
    __shared__ float  s_res[ROWS * RVQ_D];
    __shared__ __bf16 s_hi [ROWS * RVQ_D];
    __shared__ __bf16 s_lo [ROWS * RVQ_D];
    __shared__ float  s_wval[4][ROWS];
    __shared__ int    s_widx[4][ROWS];
    __shared__ int    s_fidx[ROWS];
    __shared__ float  s_sred[4];

    const int tid  = threadIdx.x;
    const int wv   = tid >> 5;
    const int lane = tid & 31;
    const int nl   = lane & 15;
    const bool hi2 = lane >= 16;
    const size_t blk = blockIdx.x;

    // ---- stage residual tile (32 rows x 256) into LDS: fp32 + bf16 hi/lo ----
    #pragma unroll
    for (int j = 0; j < (ROWS * RVQ_D) / 128; ++j) {
        int e = tid + j * 128;                 // coalesced
        float v = resin[blk * (ROWS * RVQ_D) + e];
        s_res[e] = v;
        __bf16 h = (__bf16)v;
        s_hi[e] = h;
        s_lo[e] = (__bf16)(v - (float)h);
    }
    __syncthreads();

    // ---- distance GEMM + running argmin over this wave's 16 N-tiles ----
    const size_t stageOffW = (size_t)stage * RVQ_K * RVQ_D;
    float minv0[8], minv1[8];
    int   mini0[8], mini1[8];
    #pragma unroll
    for (int v = 0; v < 8; ++v) {
        minv0[v] = 3.4e38f; mini0[v] = 0;
        minv1[v] = 3.4e38f; mini1[v] = 0;
    }

    const int aO1 = hi2 ? 16 : 0;      // A fragment K sub-offsets per lane-half
    const int aO2 = hi2 ? 24 : 8;
    const int bKo = hi2 ? 16 : 0;      // B fragment K base per lane-half

    for (int t = wv; t < 64; t += 4) {
        const int n0 = t * 16;
        v8f acc0 = {0.f, 0.f, 0.f, 0.f, 0.f, 0.f, 0.f, 0.f};
        v8f acc1 = {0.f, 0.f, 0.f, 0.f, 0.f, 0.f, 0.f, 0.f};
        const __bf16* wrh = whi + stageOffW + (size_t)(n0 + nl) * RVQ_D + bKo;
        const __bf16* wrl = wlo + stageOffW + (size_t)(n0 + nl) * RVQ_D + bKo;

        #pragma unroll
        for (int kc = 0; kc < 8; ++kc) {
            const int k0 = kc * 32;
            // B fragments (code row = n0+nl), contiguous 16 K-values per lane
            v16bf b_hi, b_lo;
            #pragma unroll
            for (int i = 0; i < 16; ++i) {
                b_hi[i] = wrh[k0 + i];
                b_lo[i] = wrl[k0 + i];
            }
            // A fragments for both 16-row tiles from LDS
            v16bf a0h = load_afrag(&s_hi[nl * RVQ_D + k0], aO1, aO2);
            v16bf a0l = load_afrag(&s_lo[nl * RVQ_D + k0], aO1, aO2);
            v16bf a1h = load_afrag(&s_hi[(16 + nl) * RVQ_D + k0], aO1, aO2);
            v16bf a1l = load_afrag(&s_lo[(16 + nl) * RVQ_D + k0], aO1, aO2);

            // r.W ~= hi.hi + hi.lo + lo.hi  (fp32 accumulate), two A-tiles
            acc0 = __builtin_amdgcn_wmma_f32_16x16x32_bf16(false, a0h, false, b_hi,
                                                           (short)0, acc0, false, false);
            acc1 = __builtin_amdgcn_wmma_f32_16x16x32_bf16(false, a1h, false, b_hi,
                                                           (short)0, acc1, false, false);
            acc0 = __builtin_amdgcn_wmma_f32_16x16x32_bf16(false, a0h, false, b_lo,
                                                           (short)0, acc0, false, false);
            acc1 = __builtin_amdgcn_wmma_f32_16x16x32_bf16(false, a1h, false, b_lo,
                                                           (short)0, acc1, false, false);
            acc0 = __builtin_amdgcn_wmma_f32_16x16x32_bf16(false, a0l, false, b_hi,
                                                           (short)0, acc0, false, false);
            acc1 = __builtin_amdgcn_wmma_f32_16x16x32_bf16(false, a1l, false, b_hi,
                                                           (short)0, acc1, false, false);
        }
        // d2 = ||W_n||^2 - 2*dot ; C layout: vgpr v -> row (v or v+8), col = nl
        const int n = n0 + nl;
        const float c2n = c2[stage * RVQ_K + n];
        #pragma unroll
        for (int v = 0; v < 8; ++v) {
            float d20 = c2n - 2.0f * acc0[v];
            float d21 = c2n - 2.0f * acc1[v];
            if (d20 < minv0[v]) { minv0[v] = d20; mini0[v] = n; }
            if (d21 < minv1[v]) { minv1[v] = d21; mini1[v] = n; }
        }
    }

    // ---- argmin reduce across the 16 lanes of each half (first-index ties) ----
    #pragma unroll
    for (int off = 8; off >= 1; off >>= 1) {
        #pragma unroll
        for (int v = 0; v < 8; ++v) {
            float ov0 = __shfl_xor(minv0[v], off, 16);
            int   oi0 = __shfl_xor(mini0[v], off, 16);
            if (ov0 < minv0[v] || (ov0 == minv0[v] && oi0 < mini0[v])) {
                minv0[v] = ov0; mini0[v] = oi0;
            }
            float ov1 = __shfl_xor(minv1[v], off, 16);
            int   oi1 = __shfl_xor(mini1[v], off, 16);
            if (ov1 < minv1[v] || (ov1 == minv1[v] && oi1 < mini1[v])) {
                minv1[v] = ov1; mini1[v] = oi1;
            }
        }
    }
    if (nl == 0) {
        const int rbase = hi2 ? 8 : 0;
        #pragma unroll
        for (int v = 0; v < 8; ++v) {
            s_wval[wv][rbase + v]      = minv0[v];
            s_widx[wv][rbase + v]      = mini0[v];
            s_wval[wv][16 + rbase + v] = minv1[v];
            s_widx[wv][16 + rbase + v] = mini1[v];
        }
    }
    __syncthreads();

    // ---- reduce across the 4 waves, emit index output ----
    if (tid < ROWS) {
        float bv = s_wval[0][tid];
        int   bi = s_widx[0][tid];
        #pragma unroll
        for (int w = 1; w < 4; ++w) {
            float v = s_wval[w][tid];
            int   i = s_widx[w][tid];
            if (v < bv || (v == bv && i < bi)) { bv = v; bi = i; }
        }
        s_fidx[tid] = bi;
        const size_t b = blk * ROWS + tid;
        out[(size_t)RVQ_B * RVQ_Q * RVQ_D + b * RVQ_Q + stage] = (float)bi;
    }
    __syncthreads();

    // ---- gather codeword, write quantized (= q_ste), update residual, sumsq ----
    float ss = 0.f;
    #pragma unroll
    for (int j = 0; j < (ROWS * RVQ_D) / 128; ++j) {
        int e   = tid + j * 128;
        int row = e >> 8;
        int d   = e & 255;
        int qi  = s_fidx[row];
        float qv = cb[stageOffW + (size_t)qi * RVQ_D + d];
        float rn = s_res[e] - qv;
        resout[blk * (ROWS * RVQ_D) + e] = rn;
        const size_t b = blk * ROWS + row;
        out[b * (RVQ_Q * RVQ_D) + (size_t)stage * RVQ_D + d] = qv;
        ss += rn * rn;
    }
    #pragma unroll
    for (int off = 16; off >= 1; off >>= 1) ss += __shfl_xor(ss, off, 32);
    if (lane == 0) s_sred[wv] = ss;
    __syncthreads();
    if (tid == 0)
        partials[stage * NBLK + blk] = s_sred[0] + s_sred[1] + s_sred[2] + s_sred[3];
}

// ---------------------------------------------------------------------------
// loss = mean(res_final^2) + 0.25 * sum_q mean(res_after_q^2)   (deterministic)
// ---------------------------------------------------------------------------
__global__ void rvq_finalize_kernel(const float* __restrict__ partials,
                                    float* __restrict__ out) {
    __shared__ float red[8][4];
    const int tid = threadIdx.x, lane = tid & 31, wv = tid >> 5;
    float loc[4] = {0.f, 0.f, 0.f, 0.f};
    for (int i = tid; i < NBLK; i += 256) {
        #pragma unroll
        for (int s = 0; s < 4; ++s) loc[s] += partials[s * NBLK + i];
    }
    #pragma unroll
    for (int off = 16; off >= 1; off >>= 1) {
        #pragma unroll
        for (int s = 0; s < 4; ++s) loc[s] += __shfl_xor(loc[s], off, 32);
    }
    if (lane == 0) {
        #pragma unroll
        for (int s = 0; s < 4; ++s) red[wv][s] = loc[s];
    }
    __syncthreads();
    if (tid == 0) {
        float ss[4];
        #pragma unroll
        for (int s = 0; s < 4; ++s) {
            ss[s] = 0.f;
            for (int w = 0; w < 8; ++w) ss[s] += red[w][s];
        }
        const float invBD = 1.0f / ((float)RVQ_B * (float)RVQ_D);
        float loss = ss[3] * invBD + 0.25f * (ss[0] + ss[1] + ss[2] + ss[3]) * invBD;
        out[(size_t)RVQ_B * RVQ_Q * RVQ_D + (size_t)RVQ_B * RVQ_Q] = loss;
    }
}

// ---------------------------------------------------------------------------
extern "C" void kernel_launch(void* const* d_in, const int* in_sizes, int n_in,
                              void* d_out, int out_size, void* d_ws, size_t ws_size,
                              hipStream_t stream) {
    const float* x  = (const float*)d_in[0];        // (B, D)
    const float* cb = (const float*)d_in[1];        // (Q, K, D)
    float* out = (float*)d_out;

    char* ws = (char*)d_ws;
    float*  res      = (float*)ws;                                   // B*D f32  (64 MB)
    size_t  off      = (size_t)RVQ_B * RVQ_D * sizeof(float);
    __bf16* whi      = (__bf16*)(ws + off); off += (size_t)RVQ_Q * RVQ_K * RVQ_D * 2;
    __bf16* wlo      = (__bf16*)(ws + off); off += (size_t)RVQ_Q * RVQ_K * RVQ_D * 2;
    float*  c2       = (float*)(ws + off);  off += (size_t)RVQ_Q * RVQ_K * sizeof(float);
    float*  partials = (float*)(ws + off);

    // codebook prep (cheap: 4 MB)
    rvq_convert_kernel<<<(RVQ_Q * RVQ_K * RVQ_D) / 256, 256, 0, stream>>>(cb, whi, wlo);
    rvq_c2_kernel<<<(RVQ_Q * RVQ_K) / 4, 128, 0, stream>>>(cb, c2);

    // sequential residual stages; stage 0 consumes x directly
    for (int stage = 0; stage < RVQ_Q; ++stage) {
        rvq_stage_kernel<<<NBLK, 128, 0, stream>>>(
            stage == 0 ? x : res, res, cb, whi, wlo, c2, out, partials, stage);
    }

    rvq_finalize_kernel<<<1, 256, 0, stream>>>(partials, out);
}
